// INSERT_17935783428478
// MI455X (gfx1250) — matrix operand order repaired and checked by reading, not compiled, not run on hardware
//
#include <hip/hip_runtime.h>
#include <hip/hip_bf16.h>

#define H 128
#define Bb 64
#define T 20
#define S 10
#define L 20
#define NITEMS 50000
#define NSEQ (Bb + Bb*S)      // 704 sequences (cur + friend)
#define M_ROWS (Bb*T)         // 1280
#define KDIM 256              // effective K of final GEMM (hu block is zero)
#define NT_TOT (NITEMS/16)    // 3125 n-tiles
#define KT_GEMM 8             // 256/32
#define KT_GRU 4              // 128/32
#define MT_TOT (M_ROWS/16)    // 80 m-tiles

typedef __attribute__((ext_vector_type(16))) __bf16 v16bf;
typedef __attribute__((ext_vector_type(8)))  float  v8f;

__device__ __forceinline__ unsigned short f2bf(float f) {
  unsigned int u = __float_as_uint(f);
  u += 0x7FFFu + ((u >> 16) & 1u);     // round-to-nearest-even
  return (unsigned short)(u >> 16);
}

__device__ __forceinline__ v16bf load_frag16(const unsigned short* p) {
  union { uint4 u[2]; v16bf v; } cv;
  const uint4* q = reinterpret_cast<const uint4*>(p);
  cv.u[0] = q[0]; cv.u[1] = q[1];
  return cv.v;
}

// ---------------------------------------------------------------------------
// Pack GRU weights W_ih/W_hh (row-major [384,128] f32) into bf16 WMMA
// B-fragment tiles: wB[(nt*4+kt)*512 + lane*16 + idx]
// ---------------------------------------------------------------------------
__global__ void pack_gru_w(const float* __restrict__ Wih,
                           const float* __restrict__ Whh,
                           unsigned short* __restrict__ wBih,
                           unsigned short* __restrict__ wBhh) {
  int e = blockIdx.x * 256 + threadIdx.x;
  if (e >= 2 * 24 * KT_GRU * 512) return;
  int mat = e / (24 * KT_GRU * 512);
  int r   = e % (24 * KT_GRU * 512);
  int nt  = r / (KT_GRU * 512);
  int r2  = r % (KT_GRU * 512);
  int kt  = r2 / 512;
  int r3  = r2 % 512;
  int kk  = r3 >> 4;            // 0..31 (K within tile)
  int n   = r3 & 15;            // 0..15 (N within tile)
  int k   = kt * 32 + kk;       // 0..127
  int j   = nt * 16 + n;        // 0..383
  const float* W = mat ? Whh : Wih;
  unsigned short v = f2bf(W[j * H + k]);   // B[k][j] = W[j][k]
  int ln = n + (((kk >> 3) & 1) << 4);
  int ix = (kk & 7) + (((kk >> 4) & 1) << 3);
  unsigned short* dst = mat ? wBhh : wBih;
  dst[((nt * KT_GRU + kt) * 32 + ln) * 16 + ix] = v;
}

// ---------------------------------------------------------------------------
// Batched WMMA GRU: block = 16 sequences, 12 waves. Per step:
//   G = [x_t ; h_t] x {W_ih^T, W_hh^T} via v_wmma_f32_16x16x32_bf16
// waves 0..7  -> gate columns [0,256)   (r,z): gi+gh accumulated together
// waves 8..11 -> gate columns [256,384) (n)  : gi and gh kept separate
// ---------------------------------------------------------------------------
__global__ void __launch_bounds__(384) gru_wmma(
    const int* __restrict__ cur_sess, const int* __restrict__ frds_sess,
    const float* __restrict__ v_emb,
    const float* __restrict__ b_ih, const float* __restrict__ b_hh,
    const unsigned short* __restrict__ wBih,
    const unsigned short* __restrict__ wBhh,
    float* __restrict__ gru_out) {
  __shared__ float G_rz[16 * 256];                       // 16 KB
  __shared__ float Gi_n[16 * 128];                       // 8 KB
  __shared__ float Gh_n[16 * 128];                       // 8 KB
  __shared__ float h_f32[16 * 128];                      // 8 KB
  __shared__ __align__(16) unsigned short h_frag[KT_GRU * 32 * 16]; // 4 KB
  __shared__ __align__(16) unsigned short x_frag[KT_GRU * 32 * 16]; // 4 KB

  const int tid  = threadIdx.x;
  const int wave = tid >> 5;
  const int lane = tid & 31;
  const int seq0 = blockIdx.x * 16;

  // Preload this wave's 2 n-tiles of both weight matrices (held for all 20 steps)
  v16bf Bi[2][KT_GRU], Bh[2][KT_GRU];
#pragma unroll
  for (int j = 0; j < 2; ++j) {
    int nt = wave * 2 + j;
#pragma unroll
    for (int kt = 0; kt < KT_GRU; ++kt) {
      Bi[j][kt] = load_frag16(wBih + ((size_t)(nt * KT_GRU + kt) * 32 + lane) * 16);
      Bh[j][kt] = load_frag16(wBhh + ((size_t)(nt * KT_GRU + kt) * 32 + lane) * 16);
    }
  }

  // h0 = 0
  for (int e = tid; e < 16 * 128; e += 384) h_f32[e] = 0.0f;
  for (int e = tid; e < KT_GRU * 32 * 16; e += 384) h_frag[e] = 0;

  for (int t = 0; t < T; ++t) {
    // gather x_t -> bf16 A-fragment layout
    for (int e = tid; e < 16 * 128; e += 384) {
      int m = e >> 7, hh = e & 127;
      int q = seq0 + m;
      int item = (q < Bb) ? cur_sess[q * T + t] : frds_sess[(q - Bb) * L + t];
      unsigned short v = f2bf(v_emb[(size_t)item * H + hh]);
      int kt = hh >> 5, kk = hh & 31;
      int ln = m + (((kk >> 3) & 1) << 4);
      int ix = (kk & 7) + (((kk >> 4) & 1) << 3);
      x_frag[(kt * 32 + ln) * 16 + ix] = v;
    }
    __syncthreads();

    if (wave < 8) {                       // r,z gates: combined gi+gh
      v8f acc[2] = {};
#pragma unroll
      for (int kt = 0; kt < KT_GRU; ++kt) {
        v16bf ax = load_frag16(&x_frag[(kt * 32 + lane) * 16]);
        v16bf ah = load_frag16(&h_frag[(kt * 32 + lane) * 16]);
#pragma unroll
        for (int j = 0; j < 2; ++j) {
          acc[j] = __builtin_amdgcn_wmma_f32_16x16x32_bf16(false, ax, false, Bi[j][kt], (short)0, acc[j], false, false);
          acc[j] = __builtin_amdgcn_wmma_f32_16x16x32_bf16(false, ah, false, Bh[j][kt], (short)0, acc[j], false, false);
        }
      }
#pragma unroll
      for (int j = 0; j < 2; ++j) {
        int cb = (wave * 2 + j) * 16;
#pragma unroll
        for (int g = 0; g < 8; ++g) {
          int row = g + ((lane >= 16) ? 8 : 0);
          G_rz[row * 256 + cb + (lane & 15)] = acc[j][g];
        }
      }
    } else {                              // n gate: keep gi / gh separate
      v8f aci[2] = {}, ach[2] = {};
#pragma unroll
      for (int kt = 0; kt < KT_GRU; ++kt) {
        v16bf ax = load_frag16(&x_frag[(kt * 32 + lane) * 16]);
        v16bf ah = load_frag16(&h_frag[(kt * 32 + lane) * 16]);
#pragma unroll
        for (int j = 0; j < 2; ++j) {
          aci[j] = __builtin_amdgcn_wmma_f32_16x16x32_bf16(false, ax, false, Bi[j][kt], (short)0, aci[j], false, false);
          ach[j] = __builtin_amdgcn_wmma_f32_16x16x32_bf16(false, ah, false, Bh[j][kt], (short)0, ach[j], false, false);
        }
      }
#pragma unroll
      for (int j = 0; j < 2; ++j) {
        int cb = (wave * 2 + j) * 16 - 256;
#pragma unroll
        for (int g = 0; g < 8; ++g) {
          int row = g + ((lane >= 16) ? 8 : 0);
          Gi_n[row * 128 + cb + (lane & 15)] = aci[j][g];
          Gh_n[row * 128 + cb + (lane & 15)] = ach[j][g];
        }
      }
    }
    __syncthreads();

    // gate math + repack h_t for next step
    for (int e = tid; e < 16 * 128; e += 384) {
      int m = e >> 7, hh = e & 127;
      float r = G_rz[m * 256 + hh] + b_ih[hh] + b_hh[hh];
      float z = G_rz[m * 256 + 128 + hh] + b_ih[128 + hh] + b_hh[128 + hh];
      r = 1.0f / (1.0f + __expf(-r));
      z = 1.0f / (1.0f + __expf(-z));
      float hn = Gh_n[m * 128 + hh] + b_hh[256 + hh];
      float nn = tanhf(Gi_n[m * 128 + hh] + b_ih[256 + hh] + r * hn);
      float ho = (1.0f - z) * nn + z * h_f32[e];
      h_f32[e] = ho;
      int kt = hh >> 5, kk = hh & 31;
      int ln = m + (((kk >> 3) & 1) << 4);
      int ix = (kk & 7) + (((kk >> 4) & 1) << 3);
      h_frag[(kt * 32 + ln) * 16 + ix] = f2bf(ho);
      gru_out[(size_t)(seq0 + m) * (T * H) + t * H + hh] = ho;
    }
    __syncthreads();
  }
}

// ---------------------------------------------------------------------------
// scores 'bth,bslh->bstl', max over l, softmax over s -> sim1[b][s][t]
// ---------------------------------------------------------------------------
__global__ void __launch_bounds__(128) scores_softmax(
    const float* __restrict__ gru_out, float* __restrict__ sim1) {
  __shared__ float wsum[4];
  int bt = blockIdx.x;
  int b = bt / T, t = bt % T;
  int tid = threadIdx.x, lane = tid & 31, w = tid >> 5;
  float q = gru_out[(size_t)b * T * H + t * H + tid];
  float sr[S];
  for (int s = 0; s < S; ++s) {
    float best = -3.4e38f;
    for (int l = 0; l < L; ++l) {
      float p = q * gru_out[(size_t)(Bb + b * S + s) * T * H + l * H + tid];
      for (int off = 16; off > 0; off >>= 1) p += __shfl_xor(p, off, 32);
      if (lane == 0) wsum[w] = p;
      __syncthreads();
      best = fmaxf(best, wsum[0] + wsum[1] + wsum[2] + wsum[3]);
      __syncthreads();
    }
    sr[s] = best;
  }
  if (tid == 0) {
    float mx = sr[0];
    for (int s = 1; s < S; ++s) mx = fmaxf(mx, sr[s]);
    float den = 0.f, ex[S];
    for (int s = 0; s < S; ++s) { ex[s] = __expf(sr[s] - mx); den += ex[s]; }
    for (int s = 0; s < S; ++s) sim1[(b * S + s) * T + t] = ex[s] / den;
  }
}

// ---------------------------------------------------------------------------
// top-2 over s per (b,t) -> preserve[b][s]
// ---------------------------------------------------------------------------
__global__ void __launch_bounds__(32) topk_preserve(
    const float* __restrict__ sim1, int* __restrict__ preserve) {
  __shared__ int mask;
  int b = blockIdx.x, tid = threadIdx.x;
  if (tid == 0) mask = 0;
  __syncthreads();
  if (tid < T) {
    float v[S];
    for (int s = 0; s < S; ++s) v[s] = sim1[(b * S + s) * T + tid];
    int i1 = 0;
    for (int s = 1; s < S; ++s) if (v[s] > v[i1]) i1 = s;
    int i2 = -1;
    for (int s = 0; s < S; ++s)
      if (s != i1 && (i2 < 0 || v[s] > v[i2])) i2 = s;
    atomicOr(&mask, (1 << i1) | (1 << i2));
  }
  __syncthreads();
  if (tid < S) preserve[b * S + tid] = (mask >> tid) & 1;
}

// ---------------------------------------------------------------------------
// friend-session attention: w = softmax_l(u.e · items), sess_emb = Σ w*items
// ---------------------------------------------------------------------------
__global__ void __launch_bounds__(128) sess_attn(
    const int* __restrict__ users, const int* __restrict__ frds_sess,
    const float* __restrict__ v_emb, const float* __restrict__ u_emb,
    float* __restrict__ sess_emb) {
  __shared__ float dots[L];
  __shared__ float wgt[L];
  __shared__ float wsum[4];
  int bs = blockIdx.x;          // b*S+s
  int b = bs / S;
  int tid = threadIdx.x, lane = tid & 31, w = tid >> 5;
  float ue = u_emb[(size_t)users[b] * H + tid];
  for (int l = 0; l < L; ++l) {
    int item = frds_sess[bs * L + l];
    float p = ue * v_emb[(size_t)item * H + tid];
    for (int off = 16; off > 0; off >>= 1) p += __shfl_xor(p, off, 32);
    if (lane == 0) wsum[w] = p;
    __syncthreads();
    if (tid == 0) dots[l] = wsum[0] + wsum[1] + wsum[2] + wsum[3];
    __syncthreads();
  }
  if (tid == 0) {
    float mx = dots[0];
    for (int l = 1; l < L; ++l) mx = fmaxf(mx, dots[l]);
    float den = 0.f;
    for (int l = 0; l < L; ++l) { wgt[l] = __expf(dots[l] - mx); den += wgt[l]; }
    for (int l = 0; l < L; ++l) wgt[l] /= den;
  }
  __syncthreads();
  float acc = 0.f;
  for (int l = 0; l < L; ++l) {
    int item = frds_sess[bs * L + l];
    acc += wgt[l] * v_emb[(size_t)item * H + tid];
  }
  sess_emb[(size_t)bs * H + tid] = acc;
}

// ---------------------------------------------------------------------------
// masked double-softmax over s + hf[b,t,:] = Σ_s s[b,s,t]*sess_emb[b,s,:]
// ---------------------------------------------------------------------------
__global__ void __launch_bounds__(256) mix_hf(
    const float* __restrict__ sim1, const int* __restrict__ preserve,
    const float* __restrict__ sess_emb, float* __restrict__ hf) {
  __shared__ float s_sh[S][T];
  int b = blockIdx.x, tid = threadIdx.x;
  if (tid < T) {
    int t = tid;
    float v[S]; int m[S];
    for (int s = 0; s < S; ++s) {
      m[s] = preserve[b * S + s];
      v[s] = sim1[(b * S + s) * T + t];
    }
    for (int rep = 0; rep < 2; ++rep) {      // masked softmax applied twice
      float mx = -3.4e38f;
      for (int s = 0; s < S; ++s) if (m[s]) mx = fmaxf(mx, v[s]);
      float den = 0.f;
      for (int s = 0; s < S; ++s) if (m[s]) { v[s] = __expf(v[s] - mx); den += v[s]; }
      for (int s = 0; s < S; ++s) v[s] = m[s] ? v[s] / den : 0.0f;
    }
    for (int s = 0; s < S; ++s) s_sh[s][t] = v[s];
  }
  __syncthreads();
  for (int e = tid; e < T * H; e += 256) {
    int t = e >> 7, hh = e & 127;
    float acc = 0.f;
    for (int s = 0; s < S; ++s) acc += s_sh[s][t] * sess_emb[(size_t)(b * S + s) * H + hh];
    hf[(size_t)(b * T + t) * H + hh] = acc;
  }
}

// ---------------------------------------------------------------------------
// pack A = [hs | hf] (1280 x 256) -> bf16 A-fragments
// ---------------------------------------------------------------------------
__global__ void packA_kernel(const float* __restrict__ gru_out,
                             const float* __restrict__ hf,
                             unsigned short* __restrict__ Apack) {
  int e = blockIdx.x * 256 + threadIdx.x;
  if (e >= M_ROWS * KDIM) return;
  int m = e >> 8, k = e & 255;
  int b = m / T, t = m % T;
  float val = (k < H) ? gru_out[(size_t)b * T * H + t * H + k]
                      : hf[(size_t)m * H + (k - H)];
  int mt = m >> 4, row = m & 15;
  int kt = k >> 5, kk = k & 31;
  int ln = row + (((kk >> 3) & 1) << 4);
  int ix = (kk & 7) + (((kk >> 4) & 1) << 3);
  Apack[((size_t)(mt * KT_GEMM + kt) * 32 + ln) * 16 + ix] = f2bf(val);
}

// ---------------------------------------------------------------------------
// pack out1_w columns [0:128) and [256:384) -> bf16 B-fragments (25.6 MB)
// ---------------------------------------------------------------------------
__global__ void packB_kernel(const float* __restrict__ out1_w,
                             unsigned short* __restrict__ Bpack) {
  size_t e = (size_t)blockIdx.x * 256 + threadIdx.x;
  if (e >= (size_t)NT_TOT * KT_GEMM * 512) return;
  int nt = (int)(e / (KT_GEMM * 512));
  int r  = (int)(e % (KT_GEMM * 512));
  int kt = r / 512;
  int r2 = r % 512;
  int kk = r2 >> 4;                     // 0..31
  int n  = r2 & 15;
  int kg = kt * 32 + kk;                // 0..255
  int kw = (kg < H) ? kg : (kg + H);    // skip zero hu block -> 256..383
  int ng = nt * 16 + n;
  float val = out1_w[(size_t)ng * (3 * H) + kw];
  int ln = n + (((kk >> 3) & 1) << 4);
  int ix = (kk & 7) + (((kk >> 4) & 1) << 3);
  Bpack[(size_t)(nt * KT_GEMM + kt) * 512 + ln * 16 + ix] = f2bf(val);
}

// ---------------------------------------------------------------------------
// final GEMM: logits[m][n] = A[m][:] · B[:][n] + bias[n]
// 8 waves/block, one 16-col strip per wave; B tiles live in VGPRs.
// A is streamed through a 2-deep LDS pipeline with
// GLOBAL_LOAD_ASYNC_TO_LDS_B128 (ASYNCcnt-tracked), overlapping the staging
// with the WMMA burst and the 256 MB output stream.
// ---------------------------------------------------------------------------
__global__ void __launch_bounds__(256) gemm_logits(
    const unsigned short* __restrict__ Apack,
    const unsigned short* __restrict__ Bpack,
    const float* __restrict__ out1_b, float* __restrict__ out) {
  __shared__ __align__(16) unsigned short a_sh[2][KT_GEMM * 512]; // 2 x 8 KB

  int tid = threadIdx.x, lane = tid & 31, wave = tid >> 5;
  int nt = blockIdx.x * 8 + wave;
  bool active = nt < NT_TOT;
  int ntl = active ? nt : 0;
  int n = ntl * 16 + (lane & 15);
  float bias = out1_b[n];

  v16bf Breg[KT_GEMM];
#pragma unroll
  for (int kt = 0; kt < KT_GEMM; ++kt)
    Breg[kt] = load_frag16(Bpack + (size_t)(ntl * KT_GEMM + kt) * 512 + lane * 16);

  // per-thread async-copy addresses: 32 B per thread per stage (256thr*32B=8KB)
  unsigned ldsA[2];
  ldsA[0] = (unsigned)(uintptr_t)(&a_sh[0][tid * 16]);
  ldsA[1] = (unsigned)(uintptr_t)(&a_sh[1][tid * 16]);

  auto issue_stage = [&](int stage) {
    const unsigned short* g = Apack + (size_t)stage * (KT_GEMM * 512) + tid * 16;
    unsigned l = ldsA[stage & 1];
    asm volatile("global_load_async_to_lds_b128 %0, %1, off"
                 :: "v"(l), "v"(g) : "memory");
    asm volatile("global_load_async_to_lds_b128 %0, %1, off offset:16"
                 :: "v"(l), "v"(g) : "memory");
  };

  issue_stage(0);
  issue_stage(1);

  for (int mt = 0; mt < MT_TOT; ++mt) {
    // wait for this stage's 2 async ops (the 2 still outstanding belong to mt+1)
    if (mt < MT_TOT - 1) asm volatile("s_wait_asynccnt 0x2" ::: "memory");
    else                 asm volatile("s_wait_asynccnt 0x0" ::: "memory");
    __syncthreads();
    const unsigned short* abuf = a_sh[mt & 1];
    if (active) {
      v8f acc = {};
#pragma unroll
      for (int kt = 0; kt < KT_GEMM; ++kt) {
        v16bf a = load_frag16(&abuf[(kt * 32 + lane) * 16]);
        acc = __builtin_amdgcn_wmma_f32_16x16x32_bf16(false, a, false, Breg[kt], (short)0, acc, false, false);
      }
#pragma unroll
      for (int g = 0; g < 8; ++g) {
        int row = g + ((lane >= 16) ? 8 : 0);
        out[(size_t)(mt * 16 + row) * NITEMS + n] = acc[g] + bias;
      }
    }
    __syncthreads();                       // everyone done reading buf[mt&1]
    if (mt + 2 < MT_TOT) issue_stage(mt + 2);  // recycle it
  }
}

// ---------------------------------------------------------------------------
extern "C" void kernel_launch(void* const* d_in, const int* in_sizes, int n_in,
                              void* d_out, int out_size, void* d_ws, size_t ws_size,
                              hipStream_t stream) {
  const int*   users    = (const int*)  d_in[0];
  const int*   cur_sess = (const int*)  d_in[1];
  /* d_in[2] hist_sess: unused by reference */
  const int*   frds     = (const int*)  d_in[3];
  /* d_in[4] cur_sess_len: unused (always T) */
  const float* v_emb    = (const float*)d_in[5];
  const float* u_emb    = (const float*)d_in[6];
  const float* W_ih     = (const float*)d_in[7];
  const float* W_hh     = (const float*)d_in[8];
  const float* b_ih     = (const float*)d_in[9];
  const float* b_hh     = (const float*)d_in[10];
  const float* out1_w   = (const float*)d_in[11];
  const float* out1_b   = (const float*)d_in[12];
  float* logits = (float*)d_out;

  char* ws = (char*)d_ws;
  size_t off = 0;
  auto alloc = [&](size_t bytes) {
    void* p = ws + off;
    off += (bytes + 255) & ~(size_t)255;
    return p;
  };
  float* gru_out  = (float*)alloc((size_t)NSEQ * T * H * 4);          // 6.9 MB
  float* sim1     = (float*)alloc((size_t)Bb * S * T * 4);
  int*   preserve = (int*)  alloc((size_t)Bb * S * 4);
  float* sess_emb = (float*)alloc((size_t)Bb * S * H * 4);
  float* hf       = (float*)alloc((size_t)Bb * T * H * 4);
  unsigned short* wBih  = (unsigned short*)alloc((size_t)24 * KT_GRU * 512 * 2);
  unsigned short* wBhh  = (unsigned short*)alloc((size_t)24 * KT_GRU * 512 * 2);
  unsigned short* Apack = (unsigned short*)alloc((size_t)MT_TOT * KT_GEMM * 512 * 2);
  unsigned short* Bpack = (unsigned short*)alloc((size_t)NT_TOT * KT_GEMM * 512 * 2); // 25.6 MB

  // B pack is independent of everything else -> launch first (overlaps in graph)
  packB_kernel<<<(int)(((size_t)NT_TOT * KT_GEMM * 512 + 255) / 256), 256, 0, stream>>>(out1_w, Bpack);
  pack_gru_w<<<(2 * 24 * KT_GRU * 512 + 255) / 256, 256, 0, stream>>>(W_ih, W_hh, wBih, wBhh);
  gru_wmma<<<NSEQ / 16, 384, 0, stream>>>(cur_sess, frds, v_emb, b_ih, b_hh, wBih, wBhh, gru_out);
  scores_softmax<<<Bb * T, 128, 0, stream>>>(gru_out, sim1);
  topk_preserve<<<Bb, 32, 0, stream>>>(sim1, preserve);
  sess_attn<<<Bb * S, 128, 0, stream>>>(users, frds, v_emb, u_emb, sess_emb);
  mix_hf<<<Bb, 256, 0, stream>>>(sim1, preserve, sess_emb, hf);
  packA_kernel<<<(M_ROWS * KDIM + 255) / 256, 256, 0, stream>>>(gru_out, hf, Apack);
  gemm_logits<<<(NT_TOT + 7) / 8, 256, 0, stream>>>(Apack, Bpack, out1_b, logits);
}
